// GraphConvolutionalBlock_21165598835006
// MI455X (gfx1250) — matrix operand order repaired and checked by reading, not compiled, not run on hardware
//
#include <hip/hip_runtime.h>
#include <hip/hip_bf16.h>
#include <math.h>

// ---- problem constants (from reference) ----
#define NNODES   50000
#define IN_DIMK  78
#define HEADS    10
#define CDIM     78
#define HCF      780           // HEADS * CDIM
#define NEDGE    400000
#define NBATCH   256
#define NEG_SLOPE 0.2f
#define EETOT    (NEDGE + NNODES)   // edges + self loops

typedef __attribute__((ext_vector_type(16))) __bf16 v16bf;
typedef __attribute__((ext_vector_type(8)))  __bf16 v8bf;
typedef __attribute__((ext_vector_type(8)))  float  v8f;

// ------------------------------------------------------------------
// Generic fills
// ------------------------------------------------------------------
__global__ void fill_f32(float* p, float v, long long n) {
    long long i = (long long)blockIdx.x * blockDim.x + threadIdx.x;
    if (i < n) p[i] = v;
}
__global__ void fill_u32(unsigned* p, unsigned v, long long n) {
    long long i = (long long)blockIdx.x * blockDim.x + threadIdx.x;
    if (i < n) p[i] = v;
}

// ------------------------------------------------------------------
// Tiled WMMA GEMM: OUT[M x NN] = A[M x K] * W[K x NN], fp32 in/out,
// bf16 multiply + f32 accumulate (v_wmma_f32_16x16x32_bf16).
//
// Block tile 64x128x32, 256 threads (8 waves, 2x4): each wave computes a
// 32x32 patch = 2x2 WMMA tiles -> 4 v_wmma per k-step, no branches in the
// MMA loop. A staged row-major, B staged TRANSPOSED (col-major) in LDS so
// both fragment reads are contiguous, 16B-aligned ds_load_b128s.
// Row pitch = 20 dwords (80B): 16B-aligned and conflict-free banking.
// ------------------------------------------------------------------
#define BM 64
#define BN 128
#define BK 32
#define LPITCH 20               // dwords per LDS row/col (32 bf16 + pad)

__device__ inline unsigned pack2_bf16(float a, float b) {
    union { __bf16 h[2]; unsigned u; } x;
    x.h[0] = (__bf16)a; x.h[1] = (__bf16)b;
    return x.u;                 // lowers to v_cvt_pk_bf16_f32
}

__device__ inline v16bf frag_a(const unsigned* Als, int m, bool hi) {
    // A 16-bit 16x32 layout: lanes 0-15 K{0-7,16-23}; lanes 16-31 K{8-15,24-31}
    const unsigned* p = Als + m * LPITCH;
    v8bf lo = *(const v8bf*)(p + (hi ? 4 : 0));    // K 0-7   / 8-15
    v8bf hh = *(const v8bf*)(p + (hi ? 12 : 8));   // K 16-23 / 24-31
    return __builtin_shufflevector(lo, hh, 0,1,2,3,4,5,6,7,8,9,10,11,12,13,14,15);
}
__device__ inline v16bf frag_b(const unsigned* Bls, int n, bool hi) {
    // B 16-bit 32x16 layout: lanes 0-15 K{0-7,16-23}; lanes 16-31 K{8-15,24-31}
    const unsigned* p = Bls + n * LPITCH;
    v8bf lo = *(const v8bf*)(p + (hi ? 4 : 0));
    v8bf hh = *(const v8bf*)(p + (hi ? 12 : 8));
    return __builtin_shufflevector(lo, hh, 0,1,2,3,4,5,6,7,8,9,10,11,12,13,14,15);
}

__global__ __launch_bounds__(256)
void wmma_gemm_bf16(const float* __restrict__ A, const float* __restrict__ W,
                    float* __restrict__ OUT, int M, int K, int NN, int gridN) {
    __shared__ unsigned Als[BM * LPITCH];   // 5120 B
    __shared__ unsigned Bls[BN * LPITCH];   // 10240 B

    const int t    = threadIdx.x;
    const int lane = t & 31;
    const int wv   = t >> 5;
    const int wm   = wv >> 2;               // 0..1
    const int wn   = wv & 3;                // 0..3
    const bool hi  = lane >= 16;
    const int l15  = lane & 15;

    const int bm = blockIdx.x / gridN, bn = blockIdx.x % gridN;
    const int mBase = bm * BM, nBase = bn * BN;

    // staging coordinates
    const int arow = t >> 2;                // 0..63
    const int akq  = (t & 3) * 8;           // 0,8,16,24
    const int brow = t >> 5;                // 0..7 (k within group)
    const int bc0  = lane * 4;              // 0..124

    v8f acc[2][2] = {};

    for (int k0 = 0; k0 < K; k0 += BK) {
        // ---- stage A tile (64 x 32) ----
        {
            const int rg = mBase + arow;
            float v[8];
            if (rg < M && k0 + akq + 7 < K) {               // fast path
                const float* pa = A + (size_t)rg * K + k0 + akq;
                float4 f0 = *(const float4*)(pa);
                float4 f1 = *(const float4*)(pa + 4);
                v[0]=f0.x; v[1]=f0.y; v[2]=f0.z; v[3]=f0.w;
                v[4]=f1.x; v[5]=f1.y; v[6]=f1.z; v[7]=f1.w;
                __builtin_prefetch(pa + BK, 0, 0);          // global_prefetch_b8
            } else {
#pragma unroll
                for (int i = 0; i < 8; ++i) {
                    int kk = k0 + akq + i;
                    v[i] = (rg < M && kk < K) ? A[(size_t)rg * K + kk] : 0.f;
                }
            }
            unsigned* dst = &Als[arow * LPITCH + (akq >> 1)];
#pragma unroll
            for (int i = 0; i < 4; ++i) dst[i] = pack2_bf16(v[2*i], v[2*i+1]);
        }
        // ---- stage B tile (32 x 128), transposed into Bls[col][k] ----
        {
            __bf16* bb = (__bf16*)Bls;
#pragma unroll
            for (int r = 0; r < 4; ++r) {
                const int kk = k0 + brow + r * 8;
                const int cg = nBase + bc0;
                if (kk < K && cg + 3 < NN) {                // fast path
                    const float* pw = W + (size_t)kk * NN + cg;
                    float4 f = *(const float4*)pw;
                    __builtin_prefetch(pw + (size_t)BK * NN, 0, 0);
                    bb[(bc0 + 0) * (2*LPITCH) + brow + r*8] = (__bf16)f.x;
                    bb[(bc0 + 1) * (2*LPITCH) + brow + r*8] = (__bf16)f.y;
                    bb[(bc0 + 2) * (2*LPITCH) + brow + r*8] = (__bf16)f.z;
                    bb[(bc0 + 3) * (2*LPITCH) + brow + r*8] = (__bf16)f.w;
                } else {
#pragma unroll
                    for (int j = 0; j < 4; ++j) {
                        float v = (kk < K && cg + j < NN) ? W[(size_t)kk * NN + cg + j] : 0.f;
                        bb[(bc0 + j) * (2*LPITCH) + brow + r*8] = (__bf16)v;
                    }
                }
            }
        }
        __syncthreads();

        // ---- 2x2 WMMA tiles per wave, branch-free ----
#pragma unroll
        for (int mt = 0; mt < 2; ++mt) {
            v16bf af = frag_a(Als, wm * 32 + mt * 16 + l15, hi);
#pragma unroll
            for (int nt = 0; nt < 2; ++nt) {
                v16bf bf = frag_b(Bls, wn * 32 + nt * 16 + l15, hi);
                acc[mt][nt] = __builtin_amdgcn_wmma_f32_16x16x32_bf16(
                    false, af, false, bf, (short)0, acc[mt][nt], false, false);
            }
        }
        __syncthreads();
    }

    // ---- store: f32 C/D layout (lanes 0-15 M=r, lanes 16-31 M=r+8) ----
#pragma unroll
    for (int mt = 0; mt < 2; ++mt) {
#pragma unroll
        for (int nt = 0; nt < 2; ++nt) {
            const int col = nBase + wn * 32 + nt * 16 + l15;
#pragma unroll
            for (int r = 0; r < 8; ++r) {
                const int row = mBase + wm * 32 + mt * 16 + (hi ? r + 8 : r);
                if (row < M && col < NN)
                    OUT[(size_t)row * NN + col] = acc[mt][nt][r];
            }
        }
    }
}

// ---- monotonic unsigned encoding for float atomicMax over any sign ----
__device__ inline unsigned enc_f32(float f) {
    unsigned u = __float_as_uint(f);
    return (u & 0x80000000u) ? ~u : (u | 0x80000000u);
}
__device__ inline float dec_f32(unsigned e) {
    unsigned u = (e & 0x80000000u) ? (e ^ 0x80000000u) : ~e;
    return __uint_as_float(u);
}
#define ENC_NEG_INF 0x007FFFFFu   // enc_f32(-inf)

// ------------------------------------------------------------------
// GATv2 attention scores + segment max: one thread per (edge, head).
// ------------------------------------------------------------------
__global__ void gat_scores(const float* __restrict__ xl, const float* __restrict__ xr,
                           const long long* __restrict__ ei, const float* __restrict__ att,
                           float* __restrict__ score, unsigned* __restrict__ m_enc) {
    long long idx = (long long)blockIdx.x * blockDim.x + threadIdx.x;
    if (idx >= (long long)EETOT * HEADS) return;
    int e = (int)(idx / HEADS), h = (int)(idx % HEADS);
    int src, dst;
    if (e < NEDGE) { src = (int)ei[e]; dst = (int)ei[NEDGE + e]; }
    else           { src = dst = e - NEDGE; }
    const float* pl = xl + (size_t)src * HCF + h * CDIM;
    const float* pr = xr + (size_t)dst * HCF + h * CDIM;
    const float* pa = att + h * CDIM;
    float s = 0.f;
    for (int c = 0; c < CDIM; ++c) {
        float v = pl[c] + pr[c];
        v = (v > 0.f) ? v : NEG_SLOPE * v;
        s = fmaf(v, pa[c], s);
    }
    score[idx] = s;
    atomicMax(&m_enc[(size_t)dst * HEADS + h], enc_f32(s));
}

// exp(score - max) in place; atomicAdd softmax denominator
__global__ void softmax_num(const long long* __restrict__ ei,
                            float* __restrict__ score,
                            const unsigned* __restrict__ m_enc,
                            float* __restrict__ denom) {
    long long idx = (long long)blockIdx.x * blockDim.x + threadIdx.x;
    if (idx >= (long long)EETOT * HEADS) return;
    int e = (int)(idx / HEADS), h = (int)(idx % HEADS);
    int dst = (e < NEDGE) ? (int)ei[NEDGE + e] : (e - NEDGE);
    float m  = dec_f32(m_enc[(size_t)dst * HEADS + h]);
    float ex = __expf(score[idx] - m);
    score[idx] = ex;
    atomicAdd(&denom[(size_t)dst * HEADS + h], ex);
}

// agg[dst,f] += (ex/denom) * xl[src,f] ; one thread per (edge, feature)
__global__ void gat_aggregate(const long long* __restrict__ ei,
                              const float* __restrict__ ex, const float* __restrict__ denom,
                              const float* __restrict__ xl, float* __restrict__ agg) {
    long long idx = (long long)blockIdx.x * blockDim.x + threadIdx.x;
    if (idx >= (long long)EETOT * HCF) return;
    int e = (int)(idx / HCF), f = (int)(idx % HCF), h = f / CDIM;
    int src, dst;
    if (e < NEDGE) { src = (int)ei[e]; dst = (int)ei[NEDGE + e]; }
    else           { src = dst = e - NEDGE; }
    float alpha = ex[(size_t)e * HEADS + h] / denom[(size_t)dst * HEADS + h];
    atomicAdd(&agg[(size_t)dst * HCF + f], alpha * xl[(size_t)src * HCF + f]);
}

// h = elu(agg + bias1), in place
__global__ void bias_elu(float* __restrict__ agg, const float* __restrict__ bias) {
    long long idx = (long long)blockIdx.x * blockDim.x + threadIdx.x;
    if (idx >= (long long)NNODES * HCF) return;
    int f = (int)(idx % HCF);
    float v = agg[idx] + bias[f];
    agg[idx] = (v > 0.f) ? v : (__expf(v) - 1.f);
}

// degree over real edges (self loops pre-seeded by fill(deg,1))
__global__ void degree_k(const long long* __restrict__ ei, float* __restrict__ deg) {
    long long e = (long long)blockIdx.x * blockDim.x + threadIdx.x;
    if (e >= NEDGE) return;
    atomicAdd(&deg[(int)ei[NEDGE + e]], 1.0f);
}
__global__ void dinv_k(const float* __restrict__ deg, float* __restrict__ dinv) {
    long long i = (long long)blockIdx.x * blockDim.x + threadIdx.x;
    if (i >= NNODES) return;
    dinv[i] = rsqrtf(fmaxf(deg[i], 1e-12f));
}

// out2[dst,f] += dinv[src]*dinv[dst]*hw[src,f]
__global__ void gcn_scatter(const long long* __restrict__ ei, const float* __restrict__ dinv,
                            const float* __restrict__ hw, float* __restrict__ out2) {
    long long idx = (long long)blockIdx.x * blockDim.x + threadIdx.x;
    if (idx >= (long long)EETOT * HCF) return;
    int e = (int)(idx / HCF), f = (int)(idx % HCF);
    int src, dst;
    if (e < NEDGE) { src = (int)ei[e]; dst = (int)ei[NEDGE + e]; }
    else           { src = dst = e - NEDGE; }
    float norm = dinv[src] * dinv[dst];
    atomicAdd(&out2[(size_t)dst * HCF + f], norm * hw[(size_t)src * HCF + f]);
}

// relu(out2 + bias_gcn) then segment max / sum / count over batch.
__global__ void pool_k(const float* __restrict__ out2, const float* __restrict__ bias,
                       const long long* __restrict__ batch,
                       unsigned* __restrict__ gmax_u, float* __restrict__ sums,
                       float* __restrict__ cnt) {
    long long idx = (long long)blockIdx.x * blockDim.x + threadIdx.x;
    if (idx >= (long long)NNODES * HCF) return;
    int i = (int)(idx / HCF), f = (int)(idx % HCF);
    float v = out2[idx] + bias[f];
    v = (v > 0.f) ? v : 0.f;
    int b = (int)batch[i];
    atomicMax(&gmax_u[(size_t)b * HCF + f], __float_as_uint(v));   // v>=0: bit order ok
    atomicAdd(&sums[(size_t)b * HCF + f], v);
    if (f == 0) atomicAdd(&cnt[b], 1.0f);
}

__global__ void writeout_k(const unsigned* __restrict__ gmax_u, const float* __restrict__ sums,
                           const float* __restrict__ cnt, float* __restrict__ out) {
    long long idx = (long long)blockIdx.x * blockDim.x + threadIdx.x;
    if (idx >= (long long)NBATCH * HCF) return;
    int b = (int)(idx / HCF), f = (int)(idx % HCF);
    out[(size_t)b * (2 * HCF) + f]       = __uint_as_float(gmax_u[idx]);
    out[(size_t)b * (2 * HCF) + HCF + f] = sums[idx] / fmaxf(cnt[b], 1.0f);
}

// ------------------------------------------------------------------
static inline int blks(long long n, int bs) { return (int)((n + bs - 1) / bs); }

extern "C" void kernel_launch(void* const* d_in, const int* in_sizes, int n_in,
                              void* d_out, int out_size, void* d_ws, size_t ws_size,
                              hipStream_t stream) {
    const float*     x       = (const float*)d_in[0];
    const long long* ei      = (const long long*)d_in[1];
    const long long* batch   = (const long long*)d_in[2];
    const float*     W_l     = (const float*)d_in[3];
    const float*     W_r     = (const float*)d_in[4];
    const float*     att     = (const float*)d_in[5];
    const float*     bias1   = (const float*)d_in[6];
    const float*     W_gcn   = (const float*)d_in[7];
    const float*     bias_g  = (const float*)d_in[8];
    float* out = (float*)d_out;

    // workspace carve-out (regions A/B reused across phases: xl->hw, xr->h->out2)
    char* base = (char*)d_ws;
    size_t off = 0;
    auto carve = [&](size_t bytes) { size_t o = off; off = (off + bytes + 255) & ~(size_t)255; return o; };
    const size_t NHC = (size_t)NNODES * HCF * sizeof(float);
    float*    bufA   = (float*)(base + carve(NHC));                       // xl -> hw
    float*    bufB   = (float*)(base + carve(NHC));                       // xr -> h -> out2
    float*    score  = (float*)(base + carve((size_t)EETOT * HEADS * 4)); // scores -> ex
    unsigned* m_enc  = (unsigned*)(base + carve((size_t)NNODES * HEADS * 4));
    float*    denom  = (float*)(base + carve((size_t)NNODES * HEADS * 4));
    float*    deg    = (float*)(base + carve((size_t)NNODES * 4));
    float*    dinv   = (float*)(base + carve((size_t)NNODES * 4));
    unsigned* gmax_u = (unsigned*)(base + carve((size_t)NBATCH * HCF * 4));
    float*    sums   = (float*)(base + carve((size_t)NBATCH * HCF * 4));
    float*    cnt    = (float*)(base + carve((size_t)NBATCH * 4));

    const int BS = 256;
    const int gridN = (HCF + BN - 1) / BN;                  // 7
    const int gridM = (NNODES + BM - 1) / BM;               // 782
    const int gemmBlocks = gridM * gridN;

    // 1) xl = x @ W_l ; xr = x @ W_r   (WMMA, bf16 mul / f32 acc)
    wmma_gemm_bf16<<<gemmBlocks, BS, 0, stream>>>(x, W_l, bufA, NNODES, IN_DIMK, HCF, gridN);
    wmma_gemm_bf16<<<gemmBlocks, BS, 0, stream>>>(x, W_r, bufB, NNODES, IN_DIMK, HCF, gridN);

    // 2) attention scores + segment softmax over dst
    const long long nEH = (long long)EETOT * HEADS;
    fill_u32<<<blks((long long)NNODES * HEADS, BS), BS, 0, stream>>>(m_enc, ENC_NEG_INF, (long long)NNODES * HEADS);
    fill_f32<<<blks((long long)NNODES * HEADS, BS), BS, 0, stream>>>(denom, 0.f, (long long)NNODES * HEADS);
    gat_scores<<<blks(nEH, BS), BS, 0, stream>>>(bufA, bufB, ei, att, score, m_enc);
    softmax_num<<<blks(nEH, BS), BS, 0, stream>>>(ei, score, m_enc, denom);

    // 3) aggregate alpha * xl[src] into agg (bufB reused; xr dead now)
    const long long nEF = (long long)EETOT * HCF;
    fill_f32<<<blks((long long)NNODES * HCF, BS), BS, 0, stream>>>(bufB, 0.f, (long long)NNODES * HCF);
    gat_aggregate<<<blks(nEF, BS), BS, 0, stream>>>(ei, score, denom, bufA, bufB);
    bias_elu<<<blks((long long)NNODES * HCF, BS), BS, 0, stream>>>(bufB, bias1);

    // 4) GCN: degree norm, hw = h @ W_gcn (WMMA), normalized scatter
    fill_f32<<<blks(NNODES, BS), BS, 0, stream>>>(deg, 1.0f, NNODES);   // self loops
    degree_k<<<blks(NEDGE, BS), BS, 0, stream>>>(ei, deg);
    dinv_k<<<blks(NNODES, BS), BS, 0, stream>>>(deg, dinv);
    wmma_gemm_bf16<<<gemmBlocks, BS, 0, stream>>>(bufB, W_gcn, bufA, NNODES, HCF, HCF, gridN);
    fill_f32<<<blks((long long)NNODES * HCF, BS), BS, 0, stream>>>(bufB, 0.f, (long long)NNODES * HCF); // out2
    gcn_scatter<<<blks(nEF, BS), BS, 0, stream>>>(ei, dinv, bufA, bufB);

    // 5) relu + per-batch max/mean pooling
    fill_u32<<<blks((long long)NBATCH * HCF, BS), BS, 0, stream>>>(gmax_u, 0u, (long long)NBATCH * HCF);
    fill_f32<<<blks((long long)NBATCH * HCF, BS), BS, 0, stream>>>(sums, 0.f, (long long)NBATCH * HCF);
    fill_f32<<<blks(NBATCH, BS), BS, 0, stream>>>(cnt, 0.f, NBATCH);
    pool_k<<<blks((long long)NNODES * HCF, BS), BS, 0, stream>>>(bufB, bias_g, batch, gmax_u, sums, cnt);
    writeout_k<<<blks((long long)NBATCH * HCF, BS), BS, 0, stream>>>(gmax_u, sums, cnt, out);
}
